// PinSage_56727928046033
// MI455X (gfx1250) — matrix operand order repaired and checked by reading, not compiled, not run on hardware
//
#include <hip/hip_runtime.h>
#include <math.h>

typedef __bf16 bf16;
typedef __attribute__((ext_vector_type(16))) __bf16 v16bf;
typedef __attribute__((ext_vector_type(8)))  float  v8f;

static constexpr int kNodes = 1000000;
static constexpr int kF  = 128;   // IN = HID = OUT = 128
static constexpr int kT  = 16;
static constexpr int kN0 = 16384;
static constexpr int kN1 = 4096;

__device__ __forceinline__ float lrelu(float x) { return x > 0.0f ? x : 0.01f * x; }

// Overlay gather: if node was rewritten by a previous layer, read the new row.
__device__ __forceinline__ const float* row_src(const float* __restrict__ h,
                                                const float* __restrict__ ovl,
                                                const int*   __restrict__ map,
                                                int idx) {
  if (map) {
    int j = map[idx];
    if (j >= 0) return ovl + (size_t)j * kF;
  }
  return h + (size_t)idx * kF;
}

// ---------------------------------------------------------------------------
// A-fragment loader: 16-bit A 16x32 VGPR layout (ISA 7.12.2).
// LDS holds A row-major bf16 [16][ldk]; K pairs are contiguous -> b32 reads.
// lanes 0-15: M=lane,    V0..3 K=0..7,  V4..7 K=16..23
// lanes16-31: M=lane-16, V0..3 K=8..15, V4..7 K=24..31
// ---------------------------------------------------------------------------
__device__ __forceinline__ v16bf load_a_frag(const unsigned* __restrict__ A32,
                                             int lane, int kbase, int ldk) {
  int m  = lane & 15;
  int hb = (lane >> 4) << 3;     // 0 or 8
  v16bf a;
#pragma unroll
  for (int j = 0; j < 8; ++j) {
    int k = kbase + hb + ((j < 4) ? (2 * j) : (2 * j + 8));
    union { unsigned u; bf16 b[2]; } cv;
    cv.u = A32[(m * ldk + k) >> 1];
    a[2 * j]     = cv.b[0];
    a[2 * j + 1] = cv.b[1];
  }
  return a;
}

// ---------------------------------------------------------------------------
// Pre-convert a weight matrix W[f32, outDim=128 x K] into bf16 B-fragments.
// Fragment layout (16-bit B 32x16, ISA 7.12.2): lane n = l%16,
// lanes 0-15 hold K=0..7,16..23, lanes 16-31 hold K=8..15,24..31 (VGPR j -> a K pair).
// pre index = (((tile*Kc + chunk)*32 + lane)*16 + e), B[k][n] = W[n*K + k].
// ---------------------------------------------------------------------------
__global__ void prep_weights(const float* __restrict__ W, bf16* __restrict__ pre, int K) {
  int p = blockIdx.x * blockDim.x + threadIdx.x;
  int Kc = K >> 5;
  int total = 8 * Kc * 32 * 16;      // 8 column tiles of 16
  if (p >= total) return;
  int e    = p & 15;
  int lane = (p >> 4) & 31;
  int c    = (p >> 9) % Kc;
  int t    = p / (Kc << 9);
  int j  = e >> 1;
  int hh = e & 1;
  int n  = (t << 4) + (lane & 15);
  int k  = (c << 5) + 2 * j + hh + ((lane >> 4) << 3) + (((2 * j + hh) >> 3) << 3);
  pre[p] = (bf16)W[n * K + k];
}

__global__ void map_init(int* __restrict__ map) {
  int i = blockIdx.x * blockDim.x + threadIdx.x;
  if (i < kNodes) map[i] = -1;
}

__global__ void map_fill(int* __restrict__ map, const int* __restrict__ ns, int n) {
  int i = blockIdx.x * blockDim.x + threadIdx.x;
  if (i < n) atomicMax(&map[ns[i]], i);   // deterministic winner for duplicates
}

// ---------------------------------------------------------------------------
// Kernel 1: per-node neighbor transform + weighted aggregate.
// One block per node; 16 neighbors form the 16-row WMMA tile; 8 waves cover
// the 128 output columns; K = 128 -> 4 bf16 WMMAs per wave.
// ---------------------------------------------------------------------------
__global__ void __launch_bounds__(256)
qagg_kernel(const float* __restrict__ h,
            const float* __restrict__ ovl, const int* __restrict__ map,
            const bf16*  __restrict__ Bpre,     // [8 tiles][4 chunks][32 lanes][16]
            const float* __restrict__ Qb,
            const int*   __restrict__ nb_nodes, // [N][16]
            const float* __restrict__ nb_w,     // [N][16]
            float* __restrict__ agg_out) {      // [N][128]
  __shared__ __align__(16) bf16 Ash[kT * kF];
  __shared__ float wsh[kT];

  const int n   = blockIdx.x;
  const int tid = threadIdx.x;

  // Gather 16 neighbor rows (f32) -> bf16 LDS tile. 16 threads per row, 8 f/thr.
  {
    int r = tid >> 4, g = tid & 15;
    int idx = nb_nodes[n * kT + r];
    const float* src = row_src(h, ovl, map, idx);
    float4 f0 = *(const float4*)(src + g * 8);
    float4 f1 = *(const float4*)(src + g * 8 + 4);
    bf16* dst = Ash + r * kF + g * 8;
    dst[0] = (bf16)f0.x; dst[1] = (bf16)f0.y; dst[2] = (bf16)f0.z; dst[3] = (bf16)f0.w;
    dst[4] = (bf16)f1.x; dst[5] = (bf16)f1.y; dst[6] = (bf16)f1.z; dst[7] = (bf16)f1.w;
  }
  if (tid < kT) wsh[tid] = nb_w[n * kT + tid];
  __syncthreads();

  const int wv = tid >> 5, lane = tid & 31;
  const unsigned* A32 = (const unsigned*)Ash;
  const v16bf* Bv = (const v16bf*)Bpre;

  v8f acc = {0.f, 0.f, 0.f, 0.f, 0.f, 0.f, 0.f, 0.f};
#pragma unroll
  for (int c = 0; c < 4; ++c) {
    v16bf a = load_a_frag(A32, lane, 32 * c, kF);
    v16bf b = Bv[(wv * 4 + c) * 32 + lane];
    acc = __builtin_amdgcn_wmma_f32_16x16x32_bf16(false, a, false, b,
                                                  (short)0, acc, false, false);
  }

  const int col   = (wv << 4) + (lane & 15);
  const float qb  = Qb[col];
  const int mbase = (lane < 16) ? 0 : 8;

  float wsum = 0.f;
#pragma unroll
  for (int t = 0; t < kT; ++t) wsum += wsh[t];

  float part = 0.f;
#pragma unroll
  for (int v = 0; v < 8; ++v) {
    float q = lrelu(acc[v] + qb);          // C layout: row = v + mbase, col fixed
    part += wsh[mbase + v] * q;
  }
  part += __shfl_xor(part, 16, 32);        // combine the two row-halves
  if (lane < 16)
    agg_out[(size_t)n * kF + col] = part / (wsum == 0.f ? 1.f : wsum);
}

// ---------------------------------------------------------------------------
// Kernel 2: h_new = normalize(lrelu(concat(self, agg) @ W^T + b)).
// 16 nodes per block; A tile 16x256 bf16 in LDS; 8 K-chunks of WMMA.
// ---------------------------------------------------------------------------
__global__ void __launch_bounds__(256)
dense_kernel(const float* __restrict__ h,
             const float* __restrict__ ovl, const int* __restrict__ map,
             const int*   __restrict__ nodeset,
             const float* __restrict__ agg,     // [N][128]
             const bf16*  __restrict__ Bpre,    // [8 tiles][8 chunks][32][16]
             const float* __restrict__ Wb,
             float* __restrict__ out_new) {     // [N][128]
  __shared__ __align__(16) bf16 Ash[16 * 256];
  __shared__ float nrm[16];

  const int tid = threadIdx.x;
  const int r = tid >> 4, g = tid & 15;
  const int i = blockIdx.x * 16 + r;

  // Build concat(self, agg) row: groups 0-7 -> self cols, 8-15 -> agg cols.
  {
    const float* src;
    int coff;
    if (g < 8) { src = row_src(h, ovl, map, nodeset[i]); coff = g * 16; }
    else       { src = agg + (size_t)i * kF + (g - 8) * 16; coff = g * 16; src -= 0; }
    const float* s = (g < 8) ? (src + coff) : src;
    bf16* dst = Ash + r * 256 + coff;
#pragma unroll
    for (int q = 0; q < 4; ++q) {
      float4 f = *(const float4*)(s + q * 4);
      dst[q * 4 + 0] = (bf16)f.x; dst[q * 4 + 1] = (bf16)f.y;
      dst[q * 4 + 2] = (bf16)f.z; dst[q * 4 + 3] = (bf16)f.w;
    }
  }
  if (tid < 16) nrm[tid] = 0.f;
  __syncthreads();

  const int wv = tid >> 5, lane = tid & 31;
  const unsigned* A32 = (const unsigned*)Ash;
  const v16bf* Bv = (const v16bf*)Bpre;

  v8f acc = {0.f, 0.f, 0.f, 0.f, 0.f, 0.f, 0.f, 0.f};
#pragma unroll
  for (int c = 0; c < 8; ++c) {
    v16bf a = load_a_frag(A32, lane, 32 * c, 256);
    v16bf b = Bv[(wv * 8 + c) * 32 + lane];
    acc = __builtin_amdgcn_wmma_f32_16x16x32_bf16(false, a, false, b,
                                                  (short)0, acc, false, false);
  }

  const int col   = (wv << 4) + (lane & 15);
  const float wb  = Wb[col];
  const int mbase = (lane < 16) ? 0 : 8;

  float val[8];
#pragma unroll
  for (int v = 0; v < 8; ++v) {
    val[v] = lrelu(acc[v] + wb);
    float s = val[v] * val[v];
#pragma unroll
    for (int d = 1; d < 16; d <<= 1) s += __shfl_xor(s, d, 32); // sum over 16 cols
    if ((lane & 15) == 0) atomicAdd(&nrm[mbase + v], s);        // ds_add_f32
  }
  __syncthreads();

#pragma unroll
  for (int v = 0; v < 8; ++v) {
    int row = mbase + v;
    float nn = sqrtf(nrm[row]);
    float inv = (nn == 0.f) ? 1.f : 1.f / nn;                   // safediv
    out_new[(size_t)(blockIdx.x * 16 + row) * kF + col] = val[v] * inv;
  }
}

// Final: out[k] = h[nodeset1][k] after scatter = new1[winner(nodeset1[k])].
__global__ void gather_out(const float* __restrict__ newv,
                           const int* __restrict__ map,
                           const int* __restrict__ ns,
                           float* __restrict__ out, int n) {
  int k = blockIdx.x * blockDim.x + threadIdx.x;
  if (k >= n * kF) return;
  int i = k >> 7, c = k & 127;
  int j = map[ns[i]];
  out[k] = newv[(size_t)j * kF + c];
}

// ---------------------------------------------------------------------------
extern "C" void kernel_launch(void* const* d_in, const int* in_sizes, int n_in,
                              void* d_out, int out_size, void* d_ws, size_t ws_size,
                              hipStream_t stream) {
  const float* h   = (const float*)d_in[0];
  const float* Qw0 = (const float*)d_in[1];
  const float* Qb0 = (const float*)d_in[2];
  const float* Ww0 = (const float*)d_in[3];
  const float* Wb0 = (const float*)d_in[4];
  const float* Qw1 = (const float*)d_in[5];
  const float* Qb1 = (const float*)d_in[6];
  const float* Ww1 = (const float*)d_in[7];
  const float* Wb1 = (const float*)d_in[8];
  const int*   ns0 = (const int*)d_in[9];
  const int*   nb0 = (const int*)d_in[10];
  const float* w0  = (const float*)d_in[11];
  const int*   ns1 = (const int*)d_in[12];
  const int*   nb1 = (const int*)d_in[13];
  const float* w1  = (const float*)d_in[14];

  // Workspace carve (all offsets 256B aligned).
  char* p = (char*)d_ws;
  auto take = [&](size_t bytes) {
    char* r = p;
    p += (bytes + 255) & ~(size_t)255;
    return r;
  };
  bf16*  Qw0p = (bf16*)take(128 * 128 * sizeof(bf16));
  bf16*  Ww0p = (bf16*)take(128 * 256 * sizeof(bf16));
  bf16*  Qw1p = (bf16*)take(128 * 128 * sizeof(bf16));
  bf16*  Ww1p = (bf16*)take(128 * 256 * sizeof(bf16));
  int*   map  = (int*)take((size_t)kNodes * sizeof(int));
  float* agg0 = (float*)take((size_t)kN0 * kF * sizeof(float));
  float* new0 = (float*)take((size_t)kN0 * kF * sizeof(float));
  float* agg1 = (float*)take((size_t)kN1 * kF * sizeof(float));
  float* new1 = (float*)take((size_t)kN1 * kF * sizeof(float));

  // Weight pre-conversion into WMMA B-fragment layout.
  prep_weights<<<64, 256, 0, stream>>>(Qw0, Qw0p, 128);
  prep_weights<<<128, 256, 0, stream>>>(Ww0, Ww0p, 256);
  prep_weights<<<64, 256, 0, stream>>>(Qw1, Qw1p, 128);
  prep_weights<<<128, 256, 0, stream>>>(Ww1, Ww1p, 256);

  // Layer 0 (reads original h).
  qagg_kernel<<<kN0, 256, 0, stream>>>(h, nullptr, nullptr, Qw0p, Qb0, nb0, w0, agg0);
  dense_kernel<<<kN0 / 16, 256, 0, stream>>>(h, nullptr, nullptr, ns0, agg0, Ww0p, Wb0, new0);

  // Sparse scatter overlay for layer 0: map node -> row of new0.
  map_init<<<(kNodes + 255) / 256, 256, 0, stream>>>(map);
  map_fill<<<(kN0 + 255) / 256, 256, 0, stream>>>(map, ns0, kN0);

  // Layer 1 (reads h overlaid with new0).
  qagg_kernel<<<kN1, 256, 0, stream>>>(h, new0, map, Qw1p, Qb1, nb1, w1, agg1);
  dense_kernel<<<kN1 / 16, 256, 0, stream>>>(h, new0, map, ns1, agg1, Ww1p, Wb1, new1);

  // Final gather h[nodeset1] after layer-1 scatter.
  map_init<<<(kNodes + 255) / 256, 256, 0, stream>>>(map);
  map_fill<<<(kN1 + 255) / 256, 256, 0, stream>>>(map, ns1, kN1);
  gather_out<<<(kN1 * kF + 255) / 256, 256, 0, stream>>>(new1, map, ns1, (float*)d_out, kN1);
}
